// Memory_35235911696939
// MI455X (gfx1250) — matrix operand order repaired and checked by reading, not compiled, not run on hardware
//
#include <hip/hip_runtime.h>
#include <cstddef>

#define NSLOTS 500000
#define BQ 256
#define KF 256
#define SLAB 128
#define KP 32
#define ASTR 48   // halves; 96B row stride (16B aligned, padded vs 64B to break bank patterns)
#define BSTR 48

typedef _Float16 h8   __attribute__((ext_vector_type(8)));
typedef _Float16 v16h __attribute__((ext_vector_type(16)));
typedef float    v8f  __attribute__((ext_vector_type(8)));

// ---- monotonic float<->uint mapping (for packed argmax/argmin keys) ----
__device__ __forceinline__ unsigned ordf(float f) {
  unsigned u = __float_as_uint(f);
  return (u & 0x80000000u) ? ~u : (u | 0x80000000u);
}
__device__ __forceinline__ float unordf(unsigned u) {
  return __uint_as_float((u & 0x80000000u) ? (u & 0x7FFFFFFFu) : ~u);
}
__device__ __forceinline__ unsigned long long shflx64(unsigned long long v, int m) {
  int lo = __shfl_xor((int)(unsigned)(v & 0xFFFFFFFFull), m, 32);
  int hi = __shfl_xor((int)(unsigned)(v >> 32), m, 32);
  return ((unsigned long long)(unsigned)hi << 32) | (unsigned)lo;
}

// ---- ws init: writer overlay = -1, histogram = 0 ----
__global__ void init_kernel(int* __restrict__ writer, int* __restrict__ hist) {
  int i = blockIdx.x * blockDim.x + threadIdx.x;
  if (i < NSLOTS) writer[i] = -1;
  if (i < 1024) hist[i] = 0;
}

// ---- nearest mem_point per query: min over N of (d2, n) packed key ----
__global__ __launch_bounds__(256) void nearest_kernel(const float* __restrict__ pts,
                                                      const float* __restrict__ mpts,
                                                      unsigned long long* __restrict__ near_key) {
  int b = blockIdx.x;
  float px = pts[b * 3 + 0], py = pts[b * 3 + 1], pz = pts[b * 3 + 2];
  unsigned long long best = ~0ull;
  for (int n = threadIdx.x; n < NSLOTS; n += 256) {
    float dx = mpts[n * 3 + 0] - px;
    float dy = mpts[n * 3 + 1] - py;
    float dz = mpts[n * 3 + 2] - pz;
    float d2 = dx * dx + dy * dy + dz * dz;
    unsigned long long key = ((unsigned long long)ordf(d2) << 32) | (unsigned)n;
    best = key < best ? key : best;   // tie -> smaller n, matching jnp.argmin
  }
  __shared__ unsigned long long sr[256];
  sr[threadIdx.x] = best;
  __syncthreads();
  for (int o = 128; o > 0; o >>= 1) {
    if (threadIdx.x < o) {
      unsigned long long v = sr[threadIdx.x + o];
      if (v < sr[threadIdx.x]) sr[threadIdx.x] = v;
    }
    __syncthreads();
  }
  if (threadIdx.x == 0) near_key[b] = sr[0];
}

// ---- usage histogram (clamped to 1024 buckets) ----
__global__ void hist_kernel(const int* __restrict__ usage, int* __restrict__ hist) {
  int i = blockIdx.x * blockDim.x + threadIdx.x;
  if (i < NSLOTS) {
    int u = usage[i];
    u = u < 0 ? 0 : (u > 1023 ? 1023 : u);
    atomicAdd(&hist[u], 1);
  }
}

// ---- find rank-256 usage threshold t and count(usage < t) ----
__global__ void threshold_kernel(const int* __restrict__ hist, int* __restrict__ tc) {
  int cum = 0, t = 1023, cl = 0;
  for (int v = 0; v < 1024; ++v) {
    int h = hist[v];
    if (cum + h >= BQ) { t = v; cl = cum; break; }
    cum += h;
  }
  tc[0] = t;
  tc[1] = cl;
}

// ---- ordered compaction: 256 lowest-(usage,index) slots ----
__global__ __launch_bounds__(256) void collect_kernel(const int* __restrict__ usage,
                                                      const int* __restrict__ tc,
                                                      int* __restrict__ low_usage) {
  __shared__ int s[256];
  __shared__ int sbase;
  int tid = threadIdx.x;
  int t = tc[0], cl = tc[1];
  for (int pass = 0; pass < 2; ++pass) {
    int limit = pass ? BQ : cl;
    if (tid == 0) sbase = pass ? cl : 0;
    __syncthreads();
    if (sbase >= limit) { __syncthreads(); continue; }
    for (int chunk = 0; chunk < NSLOTS; chunk += 256) {
      int n = chunk + tid;
      int u = (n < NSLOTS) ? usage[n] : 0x7FFFFFFF;
      int p = pass ? (u == t) : (u < t);
      s[tid] = p;
      __syncthreads();
      for (int o = 1; o < 256; o <<= 1) {
        int v = (tid >= o) ? s[tid - o] : 0;
        __syncthreads();
        s[tid] += v;
        __syncthreads();
      }
      int incl = s[tid];
      int tot = s[255];
      int wb = sbase;
      if (p) {
        int pos = wb + incl - 1;
        if (pos < limit) low_usage[pos] = n;
      }
      __syncthreads();
      if (tid == 0) sbase += tot;
      __syncthreads();
      if (sbase >= limit) break;
    }
  }
}

// ---- mask/rank/idx combine + query norms + maxkey init ----
__global__ __launch_bounds__(256) void combine_kernel(const unsigned long long* __restrict__ near_key,
                                                      const int* __restrict__ low_usage,
                                                      const float* __restrict__ desc,
                                                      int* __restrict__ idx_final,
                                                      float* __restrict__ xn,
                                                      unsigned long long* __restrict__ maxkey) {
  __shared__ int s[256];
  int b = threadIdx.x;
  unsigned long long nk = near_key[b];
  float d2 = unordf((unsigned)(nk >> 32));
  int msk = (d2 > 1e-6f) ? 1 : 0;   // mind > 1e-3  <=>  d2 > 1e-6
  s[b] = msk;
  __syncthreads();
  for (int o = 1; o < 256; o <<= 1) {
    int v = (b >= o) ? s[b - o] : 0;
    __syncthreads();
    s[b] += v;
    __syncthreads();
  }
  int rank = s[b] - 1;
  if (rank < 0) rank = 0;
  int nidx = (int)(nk & 0xFFFFFFFFull);
  idx_final[b] = msk ? low_usage[rank] : nidx;
  float acc = 0.f;
  for (int f = 0; f < KF; ++f) {
    float v = desc[(size_t)b * KF + f];
    acc += v * v;
  }
  xn[b] = sqrtf(acc);
  maxkey[b] = 0ull;
}

// ---- scatter: winner-per-slot overlay (max b == last write wins) ----
__global__ void scatter_kernel(const int* __restrict__ idx_final, int* __restrict__ writer) {
  int b = threadIdx.x;
  atomicMax(&writer[idx_final[b]], b);
}

// ---- main: f16 WMMA cosine-similarity GEMM + fused masked argmax ----
__global__ __launch_bounds__(256) void sim_kernel(const float* __restrict__ desc,
                                                  const float* __restrict__ memdesc,
                                                  const int* __restrict__ usage,
                                                  const int* __restrict__ writer,
                                                  const float* __restrict__ xn,
                                                  unsigned long long* __restrict__ maxkey) {
  __shared__ _Float16 Ald[BQ * ASTR];    // 24 KB: 256 queries x 32-K panel
  __shared__ _Float16 Bld[SLAB * BSTR];  // 12 KB: 128 slots  x 32-K panel
  __shared__ float Ssq[256];             // per-(slot,half) sumsq partials
  __shared__ int Sval[SLAB];

  const int tid = threadIdx.x;
  const int base = blockIdx.x * SLAB;

  if (tid < SLAB) {
    int n = base + tid;
    Sval[tid] = (n < NSLOTS) && ((usage[n] > 0) || (writer[n] >= 0));
  }

  // B staging assignment: thread -> (slot, 16-wide K half); overlay redirect
  const int slB = tid >> 1;
  const int khB = (tid & 1) * 16;
  const int nB = base + slB;
  const float* srcB = nullptr;
  if (nB < NSLOTS) {
    int w = writer[nB];
    srcB = (w >= 0) ? (desc + (size_t)w * KF) : (memdesc + (size_t)nB * KF);
  }
  float ssq_local = 0.0f;

  v8f acc[2][8];
#pragma unroll
  for (int bt = 0; bt < 2; ++bt)
#pragma unroll
    for (int st = 0; st < 8; ++st)
#pragma unroll
      for (int g = 0; g < 8; ++g) acc[bt][st][g] = 0.0f;

  const int wave = tid >> 5, lane = tid & 31;
  const int m = lane & 15, hf = lane >> 4;
  const int rowA0 = (wave * 2 + 0) * 16 + m;
  const int rowA1 = (wave * 2 + 1) * 16 + m;

  for (int kp = 0; kp < KF; kp += KP) {
    // ---- stage A panel (2 units of 16 contiguous K values per thread) ----
#pragma unroll
    for (int uu = 0; uu < 2; ++uu) {
      int u = tid + uu * 256;
      int row = u >> 1, kh = (u & 1) * 16;
      const float4* s4 = (const float4*)(desc + (size_t)row * KF + kp + kh);
      float4 a0 = s4[0], a1 = s4[1], a2 = s4[2], a3 = s4[3];
      h8 h0, h1;
      h0[0] = (_Float16)a0.x; h0[1] = (_Float16)a0.y; h0[2] = (_Float16)a0.z; h0[3] = (_Float16)a0.w;
      h0[4] = (_Float16)a1.x; h0[5] = (_Float16)a1.y; h0[6] = (_Float16)a1.z; h0[7] = (_Float16)a1.w;
      h1[0] = (_Float16)a2.x; h1[1] = (_Float16)a2.y; h1[2] = (_Float16)a2.z; h1[3] = (_Float16)a2.w;
      h1[4] = (_Float16)a3.x; h1[5] = (_Float16)a3.y; h1[6] = (_Float16)a3.z; h1[7] = (_Float16)a3.w;
      *(h8*)&Ald[row * ASTR + kh] = h0;
      *(h8*)&Ald[row * ASTR + kh + 8] = h1;
    }
    // ---- stage B panel (with overlay source + norm accumulation) ----
    {
      float4 b0, b1, b2, b3;
      if (srcB) {
        const float4* s4 = (const float4*)(srcB + kp + khB);
        b0 = s4[0]; b1 = s4[1]; b2 = s4[2]; b3 = s4[3];
        if (kp + KP < KF) __builtin_prefetch(srcB + kp + KP + khB, 0, 0);  // global_prefetch_b8
      } else {
        b0 = make_float4(0.f, 0.f, 0.f, 0.f); b1 = b0; b2 = b0; b3 = b0;
      }
      ssq_local += b0.x * b0.x + b0.y * b0.y + b0.z * b0.z + b0.w * b0.w;
      ssq_local += b1.x * b1.x + b1.y * b1.y + b1.z * b1.z + b1.w * b1.w;
      ssq_local += b2.x * b2.x + b2.y * b2.y + b2.z * b2.z + b2.w * b2.w;
      ssq_local += b3.x * b3.x + b3.y * b3.y + b3.z * b3.z + b3.w * b3.w;
      h8 h0, h1;
      h0[0] = (_Float16)b0.x; h0[1] = (_Float16)b0.y; h0[2] = (_Float16)b0.z; h0[3] = (_Float16)b0.w;
      h0[4] = (_Float16)b1.x; h0[5] = (_Float16)b1.y; h0[6] = (_Float16)b1.z; h0[7] = (_Float16)b1.w;
      h1[0] = (_Float16)b2.x; h1[1] = (_Float16)b2.y; h1[2] = (_Float16)b2.z; h1[3] = (_Float16)b2.w;
      h1[4] = (_Float16)b3.x; h1[5] = (_Float16)b3.y; h1[6] = (_Float16)b3.z; h1[7] = (_Float16)b3.w;
      *(h8*)&Bld[slB * BSTR + khB] = h0;
      *(h8*)&Bld[slB * BSTR + khB + 8] = h1;
    }
    __syncthreads();

    // ---- A fragments (16-bit A 16x32 layout: lanes<16 -> K 0..7 & 16..23) ----
    h8 al, ah;
    al = *(const h8*)&Ald[rowA0 * ASTR + hf * 8];
    ah = *(const h8*)&Ald[rowA0 * ASTR + 16 + hf * 8];
    v16h a0;
#pragma unroll
    for (int i = 0; i < 8; ++i) { a0[i] = al[i]; a0[i + 8] = ah[i]; }
    al = *(const h8*)&Ald[rowA1 * ASTR + hf * 8];
    ah = *(const h8*)&Ald[rowA1 * ASTR + 16 + hf * 8];
    v16h a1;
#pragma unroll
    for (int i = 0; i < 8; ++i) { a1[i] = al[i]; a1[i + 8] = ah[i]; }

    // ---- B fragments (B 32x16: lanes<16 -> K 0..7 & 16..23, lanes>=16 -> K 8..15 & 24..31) ----
#pragma unroll
    for (int st = 0; st < 8; ++st) {
      int sl = st * 16 + m;
      h8 bl = *(const h8*)&Bld[sl * BSTR + hf * 8];
      h8 bh = *(const h8*)&Bld[sl * BSTR + 16 + hf * 8];
      v16h bf;
#pragma unroll
      for (int i = 0; i < 8; ++i) { bf[i] = bl[i]; bf[i + 8] = bh[i]; }
      acc[0][st] = __builtin_amdgcn_wmma_f32_16x16x32_f16(false, a0, false, bf, (short)0,
                                                          acc[0][st], false, false);
      acc[1][st] = __builtin_amdgcn_wmma_f32_16x16x32_f16(false, a1, false, bf, (short)0,
                                                          acc[1][st], false, false);
    }
    __syncthreads();
  }

  Ssq[tid] = ssq_local;
  __syncthreads();

  // ---- epilogue: normalize, mask, packed argmax (tie -> lowest n) ----
#pragma unroll
  for (int bt = 0; bt < 2; ++bt) {
#pragma unroll
    for (int g = 0; g < 8; ++g) {
      int b = (wave * 2 + bt) * 16 + g + hf * 8;  // C layout: vgpr g -> M=g / M=8+g
      float x = xn[b];
      unsigned long long best = 0ull;
#pragma unroll
      for (int st = 0; st < 8; ++st) {
        int sl = st * 16 + m;
        int n = base + sl;
        float yn = sqrtf(Ssq[sl * 2] + Ssq[sl * 2 + 1]);
        float sim = acc[bt][st][g] / fmaxf(x * yn, 1e-7f);
        if (!Sval[sl]) sim = -3.4e38f;
        unsigned long long key =
            ((unsigned long long)ordf(sim) << 32) | (unsigned long long)(0xFFFFFFFFu - (unsigned)n);
        best = key > best ? key : best;
      }
#pragma unroll
      for (int o = 1; o < 16; o <<= 1) {
        unsigned long long oth = shflx64(best, o);
        best = oth > best ? oth : best;
      }
      if (m == 0) atomicMax(maxkey + b, best);
    }
  }
}

// ---- gather + pack [written | read_desc | cos] per query row ----
__global__ __launch_bounds__(256) void output_kernel(const float* __restrict__ desc,
                                                     const float* __restrict__ memdesc,
                                                     const int* __restrict__ idx_final,
                                                     const int* __restrict__ writer,
                                                     const unsigned long long* __restrict__ maxkey,
                                                     float* __restrict__ out) {
  int b = blockIdx.x, tid = threadIdx.x;
  int wi = idx_final[b];
  int wb = writer[wi];
  if (wb < 0) wb = b;  // safety; idx_final slots are always written
  unsigned long long key = maxkey[b];
  unsigned n2 = 0xFFFFFFFFu - (unsigned)(key & 0xFFFFFFFFull);
  float cosv = unordf((unsigned)(key >> 32));
  float* orow = out + (size_t)b * (2 * KF + 1);
  orow[tid] = desc[(size_t)wb * KF + tid];
  float rv = 0.f;
  if (n2 < NSLOTS) {
    int rb = writer[n2];
    const float* rrow = (rb >= 0) ? (desc + (size_t)rb * KF) : (memdesc + (size_t)n2 * KF);
    rv = rrow[tid];
  }
  orow[KF + tid] = rv;
  if (tid == 0) orow[2 * KF] = cosv;
}

extern "C" void kernel_launch(void* const* d_in, const int* in_sizes, int n_in,
                              void* d_out, int out_size, void* d_ws, size_t ws_size,
                              hipStream_t stream) {
  (void)in_sizes; (void)n_in; (void)out_size; (void)ws_size;
  const float* points  = (const float*)d_in[0];
  const float* desc    = (const float*)d_in[1];
  const float* mpts    = (const float*)d_in[2];
  const float* memdesc = (const float*)d_in[3];
  const int*   usage   = (const int*)d_in[4];
  float* out = (float*)d_out;

  char* ws = (char*)d_ws;
  size_t o = 0;
  auto alloc = [&](size_t bytes) -> char* {
    char* p = ws + o;
    o = (o + bytes + 255) & ~(size_t)255;
    return p;
  };
  int* writer = (int*)alloc((size_t)NSLOTS * sizeof(int));                 // ~2 MB overlay
  unsigned long long* near_key = (unsigned long long*)alloc(BQ * 8);
  unsigned long long* maxkey   = (unsigned long long*)alloc(BQ * 8);
  int* hist      = (int*)alloc(1024 * sizeof(int));
  int* tc        = (int*)alloc(256);
  int* low_usage = (int*)alloc(BQ * sizeof(int));
  int* idx_final = (int*)alloc(BQ * sizeof(int));
  float* xn      = (float*)alloc(BQ * sizeof(float));

  int nb = (NSLOTS + 255) / 256;
  init_kernel<<<nb, 256, 0, stream>>>(writer, hist);
  nearest_kernel<<<BQ, 256, 0, stream>>>(points, mpts, near_key);
  hist_kernel<<<nb, 256, 0, stream>>>(usage, hist);
  threshold_kernel<<<1, 1, 0, stream>>>(hist, tc);
  collect_kernel<<<1, 256, 0, stream>>>(usage, tc, low_usage);
  combine_kernel<<<1, 256, 0, stream>>>(near_key, low_usage, desc, idx_final, xn, maxkey);
  scatter_kernel<<<1, 256, 0, stream>>>(idx_final, writer);
  sim_kernel<<<(NSLOTS + SLAB - 1) / SLAB, 256, 0, stream>>>(desc, memdesc, usage, writer, xn, maxkey);
  output_kernel<<<BQ, 256, 0, stream>>>(desc, memdesc, idx_final, writer, maxkey, out);
}
